// DMG_7146825581012
// MI455X (gfx1250) — matrix operand order repaired and checked by reading, not compiled, not run on hardware
//
#include <hip/hip_runtime.h>
#include <hip/hip_bf16.h>

typedef __attribute__((ext_vector_type(16))) _Float16 v16h;
typedef __attribute__((ext_vector_type(8)))  _Float16 v8h;
typedef __attribute__((ext_vector_type(8)))  float    v8f;

#define BATCH 4
#define NLAB  64
#define HH    512
#define WW    512

// ---------------------------------------------------------------------------
// Phase 1: build normalized 1-D gaussians, stored TRANSPOSED in f16 so phase 2
// can feed WMMA fragments with plain aligned vector loads.
//   axis 0 (x): Bgx[b][w][n] = gx_norm[b][n][w]   (W x N row-major, B-matrix)
//   axis 1 (y): Agy[b][h][n] = gy_norm[b][n][h]   (H x N row-major, A-matrix)
// One block per (b, n, axis); 256 threads cover 512 positions (2 each).
// ---------------------------------------------------------------------------
__global__ void DMG_gauss1d_kernel(const float* __restrict__ labels,
                                   const float* __restrict__ sigma_p,
                                   _Float16* __restrict__ Bgx,
                                   _Float16* __restrict__ Agy) {
  __shared__ float red[256];
  const int axis = blockIdx.x & 1;
  const int idx  = blockIdx.x >> 1;        // 0 .. B*N-1
  const int b    = idx >> 6;               // / NLAB
  const int n    = idx & (NLAB - 1);

  const float sigma = sigma_p[0];
  const float inv   = 1.0f / (2.0f * sigma * sigma);
  const float c     = labels[(b * NLAB + n) * 2 + axis];

  const int tid = threadIdx.x;
  const float d0 = (float)tid         - c;
  const float d1 = (float)(tid + 256) - c;
  const float e0 = __expf(-d0 * d0 * inv);
  const float e1 = __expf(-d1 * d1 * inv);

  red[tid] = e0 + e1;
  __syncthreads();
  #pragma unroll
  for (int s = 128; s > 0; s >>= 1) {
    if (tid < s) red[tid] += red[tid + s];
    __syncthreads();
  }
  const float scale = 1.0f / red[0];

  _Float16* outp = (axis == 0) ? Bgx : Agy;
  outp[((size_t)b * 512 + tid)       * NLAB + n] = (_Float16)(e0 * scale);
  outp[((size_t)b * 512 + tid + 256) * NLAB + n] = (_Float16)(e1 * scale);
}

// ---------------------------------------------------------------------------
// Phase 2: density[b] = Agy[b] (512x64) @ Bgx[b]^T-view (64x512) via
// v_wmma_f32_16x16x32_f16.  One 16x16 output tile per wave, K=64 -> 2 WMMAs.
// A-fragment (16-bit 16x32): lane m=lane%16 holds row (tm*16+m);
//   lanes 0-15: K = kbase+{0..15}; lanes 16-31: K = kbase+{16..31}
//   -> two aligned 128-bit loads from row-major Agy.
// B-fragment (16-bit 32x16): lane holds column tn*16 + lane%16, 16 contiguous
//   K values (lane-half selects K 0..15 vs 16..31) -> one aligned 32-byte load
//   from row-major Bgx (whose rows are the K/label axis).
// ---------------------------------------------------------------------------
__global__ void DMG_density_wmma_kernel(const _Float16* __restrict__ Agy,
                                        const _Float16* __restrict__ Bgx,
                                        float* __restrict__ out) {
  const int wave = threadIdx.x >> 5;       // 8 waves / block (wave32)
  const int lane = threadIdx.x & 31;
  const int gw   = blockIdx.x * 8 + wave;  // 0 .. 4095
  const int b    = gw >> 10;               // / (32*32 tiles)
  const int t    = gw & 1023;
  const int tm   = t >> 5;                 // tile row   0..31
  const int tn   = t & 31;                 // tile col   0..31
  const int hl   = lane >> 4;              // lane half
  const int l16  = lane & 15;

  const _Float16* Ab = Agy + (size_t)b * 512 * NLAB;
  const _Float16* Bb = Bgx + (size_t)b * 512 * NLAB;

  v8f c = {};
  #pragma unroll
  for (int kk = 0; kk < 2; ++kk) {
    const int kbase = kk * 32;
    const _Float16* ap = Ab + ((size_t)(tm * 16 + l16)) * NLAB + kbase + hl * 16;
    v8h a0 = *(const v8h*)ap;              // K chunk 0..7 / 16..23
    v8h a1 = *(const v8h*)(ap + 8);        // K chunk 8..15 / 24..31
    v16h a = __builtin_shufflevector(a0, a1,
                                     0, 1, 2, 3, 4, 5, 6, 7,
                                     8, 9, 10, 11, 12, 13, 14, 15);
    const _Float16* bp = Bb + ((size_t)(tn * 16 + l16)) * NLAB + kbase + hl * 16;
    v16h bv = *(const v16h*)bp;            // 16 contiguous K for this column
    // (neg_a, A, neg_b, B, c_mod, C, reuse_a, reuse_b)
    c = __builtin_amdgcn_wmma_f32_16x16x32_f16(false, a, false, bv,
                                               (short)0, c, false, false);
  }

  // C/D layout: VGPR r <-> row (r + 8*lane_half), lane%16 <-> column.
  float* op = out + ((size_t)b * HH + tm * 16 + hl * 8) * WW + tn * 16 + l16;
  #pragma unroll
  for (int r = 0; r < 8; ++r) op[(size_t)r * WW] = c[r];
}

extern "C" void kernel_launch(void* const* d_in, const int* in_sizes, int n_in,
                              void* d_out, int out_size, void* d_ws, size_t ws_size,
                              hipStream_t stream) {
  // d_in[0] = batch_images (unused: only shapes matter)
  // d_in[1] = batch_labels (B, 64, 2) float32
  // d_in[2] = sigma        (1)        float32
  const float* labels = (const float*)d_in[1];
  const float* sigma  = (const float*)d_in[2];
  float* out = (float*)d_out;                     // (B, 1, H, W) f32

  _Float16* Bgx = (_Float16*)d_ws;                          // B*512*64 f16 = 256 KB
  _Float16* Agy = Bgx + (size_t)BATCH * 512 * NLAB;         // +256 KB

  DMG_gauss1d_kernel<<<BATCH * NLAB * 2, 256, 0, stream>>>(labels, sigma, Bgx, Agy);
  DMG_density_wmma_kernel<<<512, 256, 0, stream>>>(Agy, Bgx, out);
}
